// HeteroGNN_69965017252512
// MI455X (gfx1250) — matrix operand order repaired and checked by reading, not compiled, not run on hardware
//
#include <hip/hip_runtime.h>
#include <math.h>

// ---------------------------------------------------------------------------
// CDNA5 WMMA (wave32) vector types
// ---------------------------------------------------------------------------
typedef __attribute__((ext_vector_type(16))) __bf16 v16bf;
typedef __attribute__((ext_vector_type(8)))  __bf16 v8bf;
typedef __attribute__((ext_vector_type(4)))  __bf16 v4bf;
typedef __attribute__((ext_vector_type(8)))  float  v8f;

#define D_DIM 256
#define H_DIM 256

// ---------------------------------------------------------------------------
// Embedding gather: out[r,:] = table[idx[r],:]   (grid = Nn, block = 256)
// ---------------------------------------------------------------------------
__global__ __launch_bounds__(256)
void gather_rows(const float* __restrict__ table, const int* __restrict__ idx,
                 float* __restrict__ out)
{
    const int r = blockIdx.x;
    const int c = threadIdx.x;
    out[(size_t)r * D_DIM + c] = table[(size_t)idx[r] * D_DIM + c];
}

// ---------------------------------------------------------------------------
// fp32 -> bf16 bulk convert (4 elements / thread)
// ---------------------------------------------------------------------------
__global__ __launch_bounds__(256)
void cvt_bf16(const float* __restrict__ in, __bf16* __restrict__ out, long long n4)
{
    const long long i = (long long)blockIdx.x * 256 + threadIdx.x;
    if (i >= n4) return;
    const float4 v = ((const float4*)in)[i];
    v4bf o = { (__bf16)v.x, (__bf16)v.y, (__bf16)v.z, (__bf16)v.w };
    ((v4bf*)out)[i] = o;
}

// ---------------------------------------------------------------------------
// Degree count (fp32 to match reference segment_sum of ones)
// ---------------------------------------------------------------------------
__global__ __launch_bounds__(256)
void degree_k(const int* __restrict__ edst, float* __restrict__ deg, int E)
{
    const int e = blockIdx.x * 256 + threadIdx.x;
    if (e < E) atomicAdd(&deg[edst[e]], 1.0f);
}

// ---------------------------------------------------------------------------
// Scatter-sum: agg[dst] += x[src]. 4 edges / 256-thread block, 64 lanes/edge,
// float4 gather + 4 fp32 global atomics/lane. Prefetch next edge's row.
// ---------------------------------------------------------------------------
__global__ __launch_bounds__(256)
void scatter_add(const float* __restrict__ X, const int* __restrict__ esrc,
                 const int* __restrict__ edst, float* __restrict__ agg, int E)
{
    const int e = blockIdx.x * 4 + (threadIdx.x >> 6);
    if (e >= E) return;
    const int lane = threadIdx.x & 63;
    if (e + 4 < E)   // hide gather latency: global_prefetch_b8 of next row
        __builtin_prefetch(X + (size_t)esrc[e + 4] * D_DIM + lane * 4, 0, 1);
    const float4 v = ((const float4*)(X + (size_t)esrc[e] * D_DIM))[lane];
    float* dst = agg + (size_t)edst[e] * D_DIM + lane * 4;
    atomicAdd(dst + 0, v.x);
    atomicAdd(dst + 1, v.y);
    atomicAdd(dst + 2, v.z);
    atomicAdd(dst + 3, v.w);
}

// ---------------------------------------------------------------------------
// Fused mean + bf16 convert: abf[r,:] = (bf16)(agg[r,:] / max(deg[r],1))
// One float4 / thread.
// ---------------------------------------------------------------------------
__global__ __launch_bounds__(256)
void mean_div_cvt(const float* __restrict__ agg, const float* __restrict__ deg,
                  __bf16* __restrict__ abf, long long n4)
{
    const long long i = (long long)blockIdx.x * 256 + threadIdx.x;
    if (i >= n4) return;
    const int r = (int)(i >> 6);                 // (i*4)/256
    const float s = 1.0f / fmaxf(deg[r], 1.0f);
    const float4 v = ((const float4*)agg)[i];
    v4bf o = { (__bf16)(v.x * s), (__bf16)(v.y * s),
               (__bf16)(v.z * s), (__bf16)(v.w * s) };
    ((v4bf*)abf)[i] = o;
}

// ---------------------------------------------------------------------------
// WMMA GEMM: Out[M,N] = A1@W1 (+ A2@W2) + bias.  K = 256, bf16 in, f32 acc.
//
// Block = 256 threads (8 waves). Block tile = 128 rows x 64 cols:
//   wave w -> row tile mt = blockIdx.y*8 + w; 4 column tiles per wave.
// Weight panel W[k=0..255][n_blk*64 .. +64) staged in LDS (32 KB) per pass,
// reused by all 8 waves x 4 tiles -> one A fragment feeds 4 WMMAs.
//
// Fragment layouts (ISA 7.12.2, wave32 bf16):
//   A 16x32: lane l (M = l&15): VGPR chunks = K [k0+8h, +8) and [k0+16+8h, +8)
//            (h = l>>4)  -> two contiguous b128 loads from bf16 row-major A.
//   B 32x16: lane l holds column N = l&15; its 16 K values in the same order as A
//            -> 16 2-byte LDS reads down one column of the row-major panel.
//   C/D f32 16x16: elem v -> M = v + 8h, N = lane&15.
// ---------------------------------------------------------------------------
__global__ __launch_bounds__(256)
void gemm_wmma(const __bf16* __restrict__ A1, const __bf16* __restrict__ W1,
               const __bf16* __restrict__ A2, const __bf16* __restrict__ W2,
               const float* __restrict__ bias, float* __restrict__ Out,
               int M, int N)
{
    __shared__ __align__(128) __bf16 wlds[256][64];   // 32 KB weight panel

    const int tid    = threadIdx.x;
    const int wave   = tid >> 5;
    const int lane   = tid & 31;
    const int half   = lane >> 4;
    const int lm     = lane & 15;
    const int mtiles = M >> 4;
    const int mt     = blockIdx.y * 8 + wave;          // wave-uniform
    const int nblk   = blockIdx.x * 64;

    const __bf16* arow1 = A1 + (size_t)(mt * 16 + lm) * 256;
    if (A2) __builtin_prefetch(A2 + (size_t)(mt * 16 + lm) * 256, 0, 0);

    v8f acc[4] = {};
    #pragma unroll
    for (int pass = 0; pass < 2; ++pass) {
        const __bf16* A = pass ? A2 : A1;
        const __bf16* W = pass ? W2 : W1;
        if (A == nullptr) break;                       // block-uniform

        // cooperative panel fill: thread tid copies W row k=tid, 64 cols (128B)
        {
            const __bf16* src = W + (size_t)tid * N + nblk;
            #pragma unroll
            for (int c = 0; c < 64; c += 8)
                *(v8bf*)&wlds[tid][c] = *(const v8bf*)(src + c);
        }
        __syncthreads();

        if (mt < mtiles) {                             // wave-uniform guard
            const __bf16* arow = pass ? (A2 + (size_t)(mt * 16 + lm) * 256) : arow1;
            #pragma unroll
            for (int k0 = 0; k0 < 256; k0 += 32) {
                const v8bf alo = *(const v8bf*)(arow + k0 + half * 8);
                const v8bf ahi = *(const v8bf*)(arow + k0 + 16 + half * 8);
                const v16bf a  = __builtin_shufflevector(alo, ahi,
                                     0,1,2,3,4,5,6,7,8,9,10,11,12,13,14,15);
                const int krow = k0 + half * 16 + lm;
                #pragma unroll
                for (int q = 0; q < 4; ++q) {
                    v16bf b;
                    #pragma unroll
                    for (int s = 0; s < 16; ++s)
                        b[s] = wlds[k0 + (s < 8 ? half * 8 + s : 16 + half * 8 + (s - 8))][q * 16 + lm];
                    acc[q] = __builtin_amdgcn_wmma_f32_16x16x32_bf16(
                                 false, a, false, b, (short)0, acc[q], false, false);
                }
            }
        }
        __syncthreads();
    }

    if (mt < mtiles) {
        #pragma unroll
        for (int q = 0; q < 4; ++q) {
            const int col = nblk + q * 16 + lm;
            const float bb = bias ? bias[col] : 0.0f;
            #pragma unroll
            for (int v = 0; v < 8; ++v)
                Out[(size_t)(mt * 16 + v + half * 8) * N + col] = acc[q][v] + bb;
        }
    }
}

// ---------------------------------------------------------------------------
// Row L2-normalize T[row,:256], accumulate into Y (HeteroConv sum).
// ---------------------------------------------------------------------------
__global__ __launch_bounds__(256)
void norm_add(const float* __restrict__ T, float* __restrict__ Y)
{
    __shared__ float red[8];
    const int row = blockIdx.x;
    const int c   = threadIdx.x;
    const float v = T[(size_t)row * H_DIM + c];
    float s = v * v;
    #pragma unroll
    for (int off = 16; off > 0; off >>= 1)
        s += __shfl_down(s, off, 32);
    if ((c & 31) == 0) red[c >> 5] = s;
    __syncthreads();
    if (c < 8) {
        float t = red[c];
        #pragma unroll
        for (int off = 4; off > 0; off >>= 1)
            t += __shfl_down(t, off, 8);
        if (c == 0) red[0] = t;
    }
    __syncthreads();
    const float nrm = fmaxf(sqrtf(red[0]), 1e-12f);
    Y[(size_t)row * H_DIM + c] += v / nrm;
}

// ---------------------------------------------------------------------------
// x = relu(y), fused fp32 + bf16 mirror write (one float4 / thread)
// ---------------------------------------------------------------------------
__global__ __launch_bounds__(256)
void relu_copy_cvt(const float* __restrict__ Y, float* __restrict__ X,
                   __bf16* __restrict__ Xbf, long long n4)
{
    const long long i = (long long)blockIdx.x * 256 + threadIdx.x;
    if (i >= n4) return;
    float4 v = ((const float4*)Y)[i];
    v.x = fmaxf(v.x, 0.0f); v.y = fmaxf(v.y, 0.0f);
    v.z = fmaxf(v.z, 0.0f); v.w = fmaxf(v.w, 0.0f);
    ((float4*)X)[i] = v;
    v4bf o = { (__bf16)v.x, (__bf16)v.y, (__bf16)v.z, (__bf16)v.w };
    ((v4bf*)Xbf)[i] = o;
}

// ---------------------------------------------------------------------------
// Host orchestration
// ---------------------------------------------------------------------------
extern "C" void kernel_launch(void* const* d_in, const int* in_sizes, int n_in,
                              void* d_out, int out_size, void* d_ws, size_t ws_size,
                              hipStream_t stream)
{
    (void)in_sizes; (void)n_in; (void)out_size; (void)ws_size;

    // Node types: 0=reaction, 1=complex, 2=protein, 3=molecule
    const int NNODE[4] = {100000, 50000, 200000, 150000};
    const int E = 300000, NL = 3, NT = 7, OUTC = 128;
    const int SRC[7] = {2, 3, 1, 2, 3, 0, 0};
    const int DST[7] = {0, 0, 0, 1, 1, 2, 3};
    const int MAT = D_DIM * H_DIM;                 // 65536

    const int*   idx_in[4]   = {(const int*)d_in[0], (const int*)d_in[1],
                                (const int*)d_in[2], (const int*)d_in[3]};
    const float* table_in[4] = {(const float*)d_in[4], (const float*)d_in[5],
                                (const float*)d_in[6], (const float*)d_in[7]};
    const float* W_l   = (const float*)d_in[8];    // [3,7,256,256]
    const float* W_r   = (const float*)d_in[9];    // [3,7,256,256]
    const float* bvec  = (const float*)d_in[10];   // [3,7,256]
    const float* W_out = (const float*)d_in[11];   // [256,128]
    const float* b_out = (const float*)d_in[12];   // [128]
    const int*   edges[7];
    for (int t = 0; t < NT; ++t) edges[t] = (const int*)d_in[13 + t];

    // ---- workspace layout ----
    float* p = (float*)d_ws;
    float* xs[4]; float* ys[4];
    for (int i = 0; i < 4; ++i) { xs[i] = p; p += (size_t)NNODE[i] * D_DIM; }
    for (int i = 0; i < 4; ++i) { ys[i] = p; p += (size_t)NNODE[i] * D_DIM; }
    float* agg = p;  p += (size_t)200000 * D_DIM;
    float* tmp = p;  p += (size_t)200000 * D_DIM;
    float* degs[7];
    for (int t = 0; t < NT; ++t) { degs[t] = p; p += NNODE[DST[t]]; }
    __bf16* q = (__bf16*)p;                        // bf16 region (64B aligned)
    __bf16* xbf[4];
    for (int i = 0; i < 4; ++i) { xbf[i] = q; q += (size_t)NNODE[i] * D_DIM; }
    __bf16* abf   = q; q += (size_t)200000 * D_DIM;
    __bf16* wl_bf = q; q += (size_t)NL * NT * MAT;
    __bf16* wr_bf = q; q += (size_t)NL * NT * MAT;
    __bf16* wo_bf = q; q += (size_t)D_DIM * OUTC;

    // 1) bf16 weight mirrors (once)
    {
        const long long nw = (long long)NL * NT * MAT / 4;
        cvt_bf16<<<(int)((nw + 255) / 256), 256, 0, stream>>>(W_l, wl_bf, nw);
        cvt_bf16<<<(int)((nw + 255) / 256), 256, 0, stream>>>(W_r, wr_bf, nw);
        const long long no = (long long)D_DIM * OUTC / 4;
        cvt_bf16<<<(int)((no + 255) / 256), 256, 0, stream>>>(W_out, wo_bf, no);
    }

    // 2) Materialize node features (fp32 master + bf16 mirror)
    for (int i = 0; i < 4; ++i) {
        gather_rows<<<NNODE[i], 256, 0, stream>>>(table_in[i], idx_in[i], xs[i]);
        const long long n4 = (long long)NNODE[i] * D_DIM / 4;
        cvt_bf16<<<(int)((n4 + 255) / 256), 256, 0, stream>>>(xs[i], xbf[i], n4);
    }

    // 3) Degrees (edge structure only; once)
    for (int t = 0; t < NT; ++t) {
        hipMemsetAsync(degs[t], 0, (size_t)NNODE[DST[t]] * sizeof(float), stream);
        degree_k<<<(E + 255) / 256, 256, 0, stream>>>(edges[t] + E, degs[t], E);
    }

    // 4) Layers
    for (int l = 0; l < NL; ++l) {
        for (int i = 0; i < 4; ++i)
            hipMemsetAsync(ys[i], 0, (size_t)NNODE[i] * D_DIM * sizeof(float), stream);

        for (int t = 0; t < NT; ++t) {
            const int s = SRC[t], d = DST[t], Nd = NNODE[d];
            const int* esrc = edges[t];
            const int* edst = edges[t] + E;

            hipMemsetAsync(agg, 0, (size_t)Nd * D_DIM * sizeof(float), stream);
            scatter_add<<<(E + 3) / 4, 256, 0, stream>>>(xs[s], esrc, edst, agg, E);

            const long long n4 = (long long)Nd * D_DIM / 4;
            mean_div_cvt<<<(int)((n4 + 255) / 256), 256, 0, stream>>>(agg, degs[t],
                                                                      abf, n4);

            const __bf16* Wl = wl_bf + (size_t)(l * NT + t) * MAT;
            const __bf16* Wr = wr_bf + (size_t)(l * NT + t) * MAT;
            const float*  bb = bvec + (size_t)(l * NT + t) * H_DIM;
            dim3 grid(H_DIM / 64, (Nd / 16 + 7) / 8);
            gemm_wmma<<<grid, 256, 0, stream>>>(abf, Wl, xbf[d], Wr, bb,
                                                tmp, Nd, H_DIM);

            norm_add<<<Nd, 256, 0, stream>>>(tmp, ys[d]);
        }

        for (int i = 0; i < 4; ++i) {
            const long long n4 = (long long)NNODE[i] * D_DIM / 4;
            relu_copy_cvt<<<(int)((n4 + 255) / 256), 256, 0, stream>>>(ys[i], xs[i],
                                                                       xbf[i], n4);
        }
    }

    // 5) Final projection: reaction @ W_out + b_out -> [100000, 128]
    {
        const int M = NNODE[0];
        dim3 grid(OUTC / 64, (M / 16 + 7) / 8);
        gemm_wmma<<<grid, 256, 0, stream>>>(xbf[0], wo_bf, nullptr, nullptr,
                                            b_out, (float*)d_out, M, OUTC);
    }
}
